// BasicAttention_81174881894981
// MI455X (gfx1250) — compile-verified, matrix operand order checked by reading, not run on hardware
//
#include <hip/hip_runtime.h>

typedef __attribute__((ext_vector_type(16))) _Float16 v16h;
typedef __attribute__((ext_vector_type(8)))  _Float16 v8h;
typedef __attribute__((ext_vector_type(8)))  float    v8f;

#define S_TOTAL 4096
#define DHEAD   64
#define NH      4
#define NB      4
#define C_IN    256
// fold 1/sqrt(64) and log2(e) into Q so softmax can use exp2
#define QSCALE  (0.125f * 1.44269504088896340736f)

// ---------------------------------------------------------------------------
// Fragment loader for CDNA5 WMMA f16 operands (A 16x32 MxK, or B 32x16 KxN
// loaded from an [N][K] row-major buffer -- both have identical lane layout):
// lane l -> row (l&15); halves 0..7 = k [kb..kb+7], halves 8..15 = k [kb+16..kb+23]
// with kb = (l>>4)*8.  Two 16B vector loads per fragment.
// ---------------------------------------------------------------------------
__device__ __forceinline__ v16h load_frag16x32(const _Float16* base, int stride) {
  int lane = threadIdx.x & 31;
  const _Float16* p = base + (lane & 15) * stride + ((lane >> 4) << 3);
  v8h lo = *(const v8h*)(p);
  v8h hi = *(const v8h*)(p + 16);
  v16h r;
#pragma unroll
  for (int i = 0; i < 8; ++i) { r[i] = lo[i]; r[i + 8] = hi[i]; }
  return r;
}

__device__ __forceinline__ v8f wmma_f16(v16h a, v16h b, v8f c) {
  return __builtin_amdgcn_wmma_f32_16x16x32_f16(false, a, false, b, (short)0, c,
                                                false, false);
}

// Async DMA: 16B global -> LDS, tracked by ASYNCcnt (GLOBAL_LOAD_ASYNC_TO_LDS_B128).
// LDS operand takes the workgroup-relative byte offset; for the flat-LDS
// aperture that is addr[31:0] of the generic pointer (ISA 10.2).
__device__ __forceinline__ void async_load16(const _Float16* gsrc, _Float16* ldst) {
  unsigned int l = (unsigned int)(unsigned long long)ldst;
  unsigned long long g = (unsigned long long)gsrc;
  asm volatile("global_load_async_to_lds_b128 %0, %1, off"
               :: "v"(l), "v"(g) : "memory");
}
#define S_WAIT_ASYNCCNT(n) asm volatile("s_wait_asynccnt " #n ::: "memory")

// ---------------------------------------------------------------------------
// Kernel 1: convert pointwise weight [768][256] f32 -> f16
// ---------------------------------------------------------------------------
__global__ void cvt_wpw_kernel(const float* __restrict__ w_pw,
                               _Float16* __restrict__ wph) {
  int idx = blockIdx.x * 256 + threadIdx.x;   // grid 768*256 == 196608 exact
  wph[idx] = (_Float16)w_pw[idx];
}

// ---------------------------------------------------------------------------
// Kernel 2: depthwise 3x3 conv (pad 1).  Output transposed: dwT [b][s][c] f16
// block = (hh, b); thread = channel c; loop over ww -> coalesced f16 stores.
// x row lines are L0/L2 resident across the ww loop.
// ---------------------------------------------------------------------------
__global__ void dwconv_kernel(const float* __restrict__ x,
                              const float* __restrict__ w_dw,
                              _Float16* __restrict__ dwT) {
  int hh = blockIdx.x, b = blockIdx.y, c = threadIdx.x;
  float w[9];
#pragma unroll
  for (int j = 0; j < 9; ++j) w[j] = w_dw[c * 9 + j];
  const float* xc = x + ((size_t)(b * C_IN + c)) * S_TOTAL;     // [64][64]
  _Float16* dst = dwT + ((size_t)(b * S_TOTAL + hh * 64)) * C_IN + c;
  for (int ww = 0; ww < 64; ++ww) {
    float s = 0.f;
#pragma unroll
    for (int ky = 0; ky < 3; ++ky) {
      int y = hh + ky - 1;
      if (y < 0 || y >= 64) continue;
#pragma unroll
      for (int kx = 0; kx < 3; ++kx) {
        int xx = ww + kx - 1;
        if (xx < 0 || xx >= 64) continue;
        s += xc[y * 64 + xx] * w[ky * 3 + kx];
      }
    }
    dst[(size_t)ww * C_IN] = (_Float16)s;
  }
}

// ---------------------------------------------------------------------------
// Kernel 3: projection for Q and K.  qkvT[s][o] = sum_c dwT[s][c]*w_pw[o][c]
// A = dwT rows (s), B = w_pw rows (o) -- both row-major [*][c].
// Each wave: one 16x16 tile, K-loop 256 = 8 WMMAs.  Q gets QSCALE folded in.
// Writes Q,K as [bh][s][d] f16 (coalesced: lanes advance d).
// ---------------------------------------------------------------------------
__global__ __launch_bounds__(256) void proj_qk_kernel(
    const _Float16* __restrict__ dwT, const _Float16* __restrict__ wph,
    _Float16* __restrict__ Qg, _Float16* __restrict__ Kg) {
  int wave = threadIdx.x >> 5, lane = threadIdx.x & 31;
  int s0 = blockIdx.x * 16;
  int o0 = (blockIdx.y * 8 + wave) * 16;           // o in [0,512)
  int b = blockIdx.z;
  const _Float16* abase = dwT + ((size_t)(b * S_TOTAL + s0)) * C_IN;
  const _Float16* bbase = wph + (size_t)o0 * C_IN;
  v8f acc = {};
#pragma unroll
  for (int kc = 0; kc < C_IN; kc += 32) {
    v16h a  = load_frag16x32(abase + kc, C_IN);
    v16h bb = load_frag16x32(bbase + kc, C_IN);
    acc = wmma_f16(a, bb, acc);
  }
  int o = o0 + (lane & 15);
  int qi = o >> 8;                 // 0 => Q, 1 => K
  int h = (o >> 6) & 3;
  int d = o & 63;
  _Float16* dst = (qi ? Kg : Qg) + ((size_t)(b * NH + h) * S_TOTAL) * DHEAD + d;
  float sc = qi ? 1.0f : QSCALE;
#pragma unroll
  for (int r = 0; r < 8; ++r) {
    int s = s0 + r + ((lane >> 4) << 3);
    dst[(size_t)s * DHEAD] = (_Float16)(acc[r] * sc);
  }
}

// ---------------------------------------------------------------------------
// Kernel 4: projection for V, un-transposed orientation:
// qkv[o][s] = sum_c w_pw[o][c]*dwT[s][c];  A = w_pw rows (o), B = dwT rows (s).
// Writes V transposed: VT [bh][d][t=s] f16 (coalesced: lanes advance s).
// ---------------------------------------------------------------------------
__global__ __launch_bounds__(256) void proj_v_kernel(
    const _Float16* __restrict__ dwT, const _Float16* __restrict__ wph,
    _Float16* __restrict__ VTg) {
  int wave = threadIdx.x >> 5, lane = threadIdx.x & 31;
  int s0 = (blockIdx.x * 8 + wave) * 16;
  int o0 = 512 + blockIdx.y * 16;                  // o in [512,768)
  int b = blockIdx.z;
  const _Float16* abase = wph + (size_t)o0 * C_IN;
  const _Float16* bbase = dwT + ((size_t)(b * S_TOTAL + s0)) * C_IN;
  v8f acc = {};
#pragma unroll
  for (int kc = 0; kc < C_IN; kc += 32) {
    v16h a  = load_frag16x32(abase + kc, C_IN);
    v16h bb = load_frag16x32(bbase + kc, C_IN);
    acc = wmma_f16(a, bb, acc);
  }
  int s = s0 + (lane & 15);
#pragma unroll
  for (int r = 0; r < 8; ++r) {
    int o = o0 + r + ((lane >> 4) << 3);
    int h = (o - 512) >> 6;
    int d = (o - 512) & 63;
    VTg[((size_t)(b * NH + h) * DHEAD + d) * S_TOTAL + s] = (_Float16)acc[r];
  }
}

// ---------------------------------------------------------------------------
// Kernel 5: flash attention, transposed formulation, double-buffered async
// global->LDS DMA (ASYNCcnt).
// Per wave: 16 s-columns.  S'[t,s] = K·Q^T tiles (t rows in VGPRs, s across
// lanes) -> online softmax is per-lane over VGPRs + one shfl_xor(16).
// P' C-fragment re-packs directly into the B-fragment for O^T = V^T · P'
// (half h<8 <- frag0[h], half h>=8 <- frag1[h-8]) -- no transpose traffic.
// ---------------------------------------------------------------------------
#define KT_ELEMS (32 * 64)
#define VT_ELEMS (64 * 32)
#define N_ITERS  (S_TOTAL / 32)

__global__ __launch_bounds__(256) void attn_kernel(
    const _Float16* __restrict__ Qg, const _Float16* __restrict__ Kg,
    const _Float16* __restrict__ VTg, float* __restrict__ out) {
  __shared__ __align__(16) _Float16 lds_k[2 * KT_ELEMS];   // [buf][t][d]
  __shared__ __align__(16) _Float16 lds_vt[2 * VT_ELEMS];  // [buf][d][t]
  int tid = threadIdx.x, lane = tid & 31, wave = tid >> 5;
  int bh = blockIdx.y;
  int scol0 = blockIdx.x * 128 + wave * 16;

  const _Float16* Qb = Qg + ((size_t)bh * S_TOTAL + scol0) * DHEAD;
  const _Float16* Kb = Kg + (size_t)bh * S_TOTAL * DHEAD;
  const _Float16* Vb = VTg + (size_t)bh * DHEAD * S_TOTAL;

  // Q B-fragments, loaded once (lane -> column s, halves -> d)
  v16h bq0 = load_frag16x32(Qb, DHEAD);
  v16h bq1 = load_frag16x32(Qb + 32, DHEAD);

  v8f acc0 = {}, acc1 = {}, acc2 = {}, acc3 = {};
  float m_run = -3.0e38f, l_run = 0.f;
  int vd = tid >> 2, vp = tid & 3;                    // V^T tile load mapping

  // prologue: async-load tile 0 into buffer 0 (16B per thread per tile)
  async_load16(Kb + (size_t)tid * 8, lds_k + tid * 8);
  async_load16(Vb + (size_t)vd * S_TOTAL + vp * 8, lds_vt + tid * 8);

  for (int i = 0; i < N_ITERS; ++i) {
    int t0 = i * 32;
    const _Float16* kcur = lds_k + (i & 1) * KT_ELEMS;
    const _Float16* vcur = lds_vt + (i & 1) * VT_ELEMS;
    if (i + 1 < N_ITERS) {
      // issue next tile's DMA into the other buffer; it overlaps this
      // iteration's WMMAs.  Waiting ASYNCcnt<=2 releases only tile i
      // (async loads complete in order).
      _Float16* knxt = lds_k + ((i + 1) & 1) * KT_ELEMS;
      _Float16* vnxt = lds_vt + ((i + 1) & 1) * VT_ELEMS;
      async_load16(Kb + (size_t)(t0 + 32) * DHEAD + tid * 8, knxt + tid * 8);
      async_load16(Vb + (size_t)vd * S_TOTAL + (t0 + 32) + vp * 8,
                   vnxt + tid * 8);
      S_WAIT_ASYNCCNT(2);
    } else {
      S_WAIT_ASYNCCNT(0);
    }
    __syncthreads();   // all waves' DMA for tile i landed

    // scores S'[t, s]: 2 t-subtiles x (d=0..31, d=32..63)
    v8f c0 = {}, c1 = {};
    c0 = wmma_f16(load_frag16x32(kcur + 0, 64), bq0, c0);
    c0 = wmma_f16(load_frag16x32(kcur + 32, 64), bq1, c0);
    c1 = wmma_f16(load_frag16x32(kcur + 16 * 64, 64), bq0, c1);
    c1 = wmma_f16(load_frag16x32(kcur + 16 * 64 + 32, 64), bq1, c1);

    // online softmax over t (rows): in-lane over VGPRs + shfl_xor(16)
    float tm = c0[0];
#pragma unroll
    for (int r = 0; r < 8; ++r) { tm = fmaxf(tm, c0[r]); tm = fmaxf(tm, c1[r]); }
    tm = fmaxf(tm, __shfl_xor(tm, 16, 32));
    float mn = fmaxf(m_run, tm);
    float alpha = exp2f(m_run - mn);
    float ts = 0.f;
#pragma unroll
    for (int r = 0; r < 8; ++r) {
      c0[r] = exp2f(c0[r] - mn);
      c1[r] = exp2f(c1[r] - mn);
      ts += c0[r] + c1[r];
    }
    ts += __shfl_xor(ts, 16, 32);
    l_run = l_run * alpha + ts;
    m_run = mn;
#pragma unroll
    for (int r = 0; r < 8; ++r) {
      acc0[r] *= alpha; acc1[r] *= alpha; acc2[r] *= alpha; acc3[r] *= alpha;
    }

    // P' C-fragment -> B-fragment for O^T = V^T * P'  (layout identity)
    v16h bp;
#pragma unroll
    for (int r = 0; r < 8; ++r) {
      bp[r] = (_Float16)c0[r];
      bp[r + 8] = (_Float16)c1[r];
    }
    acc0 = wmma_f16(load_frag16x32(vcur + 0 * 16 * 32, 32), bp, acc0);
    acc1 = wmma_f16(load_frag16x32(vcur + 1 * 16 * 32, 32), bp, acc1);
    acc2 = wmma_f16(load_frag16x32(vcur + 2 * 16 * 32, 32), bp, acc2);
    acc3 = wmma_f16(load_frag16x32(vcur + 3 * 16 * 32, 32), bp, acc3);
    __syncthreads();   // tile i fully consumed before i+2's DMA reuses buffer
  }

  // epilogue: out[bh*64 + d][s] = acc / l   (lanes advance s -> coalesced)
  float inv = 1.0f / l_run;
  int s = scol0 + (lane & 15);
  int rbase = (lane >> 4) << 3;
#pragma unroll
  for (int r = 0; r < 8; ++r) {
    int d = r + rbase;
    out[(size_t)(bh * 64 + d +  0) * S_TOTAL + s] = acc0[r] * inv;
    out[(size_t)(bh * 64 + d + 16) * S_TOTAL + s] = acc1[r] * inv;
    out[(size_t)(bh * 64 + d + 32) * S_TOTAL + s] = acc2[r] * inv;
    out[(size_t)(bh * 64 + d + 48) * S_TOTAL + s] = acc3[r] * inv;
  }
}

// ---------------------------------------------------------------------------
extern "C" void kernel_launch(void* const* d_in, const int* in_sizes, int n_in,
                              void* d_out, int out_size, void* d_ws,
                              size_t ws_size, hipStream_t stream) {
  (void)in_sizes; (void)n_in; (void)out_size; (void)ws_size;
  const float* x    = (const float*)d_in[0];   // [4,256,64,64]
  const float* w_dw = (const float*)d_in[1];   // [256,1,3,3]
  const float* w_pw = (const float*)d_in[2];   // [768,256]
  float* out = (float*)d_out;                  // [4,256,64,64]

  char* ws = (char*)d_ws;
  _Float16* dwT = (_Float16*)(ws);                   // 4*4096*256 f16 = 8 MiB
  _Float16* Qg  = (_Float16*)(ws + (8u  << 20));     // 16*4096*64 f16 = 8 MiB
  _Float16* Kg  = (_Float16*)(ws + (16u << 20));     // 8 MiB
  _Float16* VTg = (_Float16*)(ws + (24u << 20));     // 8 MiB
  _Float16* wph = (_Float16*)(ws + (32u << 20));     // 768*256 f16

  cvt_wpw_kernel<<<768, 256, 0, stream>>>(w_pw, wph);
  dwconv_kernel<<<dim3(64, 4), 256, 0, stream>>>(x, w_dw, dwT);
  proj_qk_kernel<<<dim3(256, 4, 4), 256, 0, stream>>>(dwT, wph, Qg, Kg);
  proj_v_kernel<<<dim3(32, 16, 4), 256, 0, stream>>>(dwT, wph, VTg);
  attn_kernel<<<dim3(32, 16), 256, 0, stream>>>(Qg, Kg, VTg, out);
}
